// AxialMultiheadAttention_52510270161044
// MI455X (gfx1250) — matrix operand order repaired and checked by reading, compiled, not run on hardware
//
#include <hip/hip_runtime.h>
#include <hip/hip_bf16.h>

// ---------------------------------------------------------------------------
// AxialMultiheadAttention (RoPE MHA) for gfx1250, bf16 WMMA pipeline.
// N=8, L=1024, D=1024, H=16, hd=64, SCALE = 1/8.
// ---------------------------------------------------------------------------

typedef __attribute__((ext_vector_type(16))) __bf16        v16bf;
typedef __attribute__((ext_vector_type(8)))  float         v8f;
typedef __attribute__((ext_vector_type(8)))  unsigned int  v8u;

#define NB   8
#define LL   1024
#define DD   1024
#define HH   16
#define HD   64
#define SCL  0.125f

__device__ __forceinline__ unsigned short f2bf_bits(float f) {
    unsigned int u = __float_as_uint(f);
    u += 0x7FFFu + ((u >> 16) & 1u);          // round-to-nearest-even
    return (unsigned short)(u >> 16);
}

__device__ __forceinline__ v8f wmma_bf16(v16bf a, v16bf b, v8f c) {
    return __builtin_amdgcn_wmma_f32_16x16x32_bf16(
        false, a, false, b, (short)0, c, false, false);
}

// -------------------------- elementwise helpers ----------------------------

__global__ void cvt_f32_bf16_kernel(const float* __restrict__ in,
                                    unsigned short* __restrict__ out,
                                    size_t n) {
    size_t i = (size_t)blockIdx.x * blockDim.x + threadIdx.x;
    if (i < n) out[i] = f2bf_bits(in[i]);
}

__global__ void zero_f32_kernel(float* __restrict__ p, size_t n) {
    size_t i = (size_t)blockIdx.x * blockDim.x + threadIdx.x;
    if (i < n) p[i] = 0.0f;
}

// ------------------------------ WMMA GEMM ----------------------------------
// C(M,N) = A(M,K) @ B(N,K)^T + bias.  A,B bf16 (ushort bits), C f32.
// Block tile 128x128, 8 waves (2x4), each wave 64x32 (4x2 16x16 tiles), K-step 32.

__global__ __launch_bounds__(256) void gemm_bf16_bias_kernel(
    const unsigned short* __restrict__ A,
    const unsigned short* __restrict__ B,
    const float* __restrict__ bias,
    float* __restrict__ C,
    int M, int N, int K)
{
    __shared__ unsigned int ldsA[128 * 16];   // 128 rows x 32 bf16 (16 uints)
    __shared__ unsigned int ldsB[128 * 16];

    const int t    = threadIdx.x;
    const int lane = t & 31;
    const int wave = t >> 5;
    const int wm   = wave >> 2;               // 0..1  -> M offset 64*wm
    const int wn   = wave & 3;                // 0..3  -> N offset 32*wn
    const int m0   = blockIdx.y * 128;
    const int n0   = blockIdx.x * 128;
    const int r    = lane & 15;
    const int hi   = lane >> 4;

    const unsigned int* Au = (const unsigned int*)A;
    const unsigned int* Bu = (const unsigned int*)B;
    const int Ku = K >> 1;                    // uints per row

    v8f acc[4][2];
    #pragma unroll
    for (int mt = 0; mt < 4; ++mt)
        #pragma unroll
        for (int nt = 0; nt < 2; ++nt)
            acc[mt][nt] = {};

    for (int k0 = 0; k0 < K; k0 += 32) {
        __syncthreads();
        #pragma unroll
        for (int i = 0; i < 8; ++i) {
            int idx = t + 256 * i;            // 0..2047
            int row = idx >> 4;
            int cu  = idx & 15;
            ldsA[idx] = Au[(size_t)(m0 + row) * Ku + (k0 >> 1) + cu];
            ldsB[idx] = Bu[(size_t)(n0 + row) * Ku + (k0 >> 1) + cu];
        }
        __syncthreads();

        v16bf bfrag[2];
        #pragma unroll
        for (int nt = 0; nt < 2; ++nt) {
            v8u bu;
            #pragma unroll
            for (int v = 0; v < 8; ++v)
                bu[v] = ldsB[(wn * 32 + nt * 16 + r) * 16 + v + 8 * hi];
            bfrag[nt] = __builtin_bit_cast(v16bf, bu);
        }
        #pragma unroll
        for (int mt = 0; mt < 4; ++mt) {
            v8u au;
            #pragma unroll
            for (int v = 0; v < 8; ++v) {
                int k2u = v + 8 * hi;
                au[v] = ldsA[(wm * 64 + mt * 16 + r) * 16 + k2u];
            }
            v16bf afrag = __builtin_bit_cast(v16bf, au);
            #pragma unroll
            for (int nt = 0; nt < 2; ++nt)
                acc[mt][nt] = wmma_bf16(afrag, bfrag[nt], acc[mt][nt]);
        }
    }

    #pragma unroll
    for (int mt = 0; mt < 4; ++mt)
        #pragma unroll
        for (int nt = 0; nt < 2; ++nt)
            #pragma unroll
            for (int v = 0; v < 8; ++v) {
                int row = m0 + wm * 64 + mt * 16 + v + 8 * hi;
                int col = n0 + wn * 32 + nt * 16 + r;
                C[(size_t)row * N + col] = acc[mt][nt][v] + bias[col];
            }
}

// ----------------------- RoPE + head split + V transpose -------------------
// qkv (N*L, 3D) f32 -> Qh,Kh (N,H,L,hd) bf16 (Q pre-scaled), Vt (N,H,hd,L) bf16.

__global__ void rope_split_kernel(const float* __restrict__ qkv,
                                  unsigned short* __restrict__ Qh,
                                  unsigned short* __restrict__ Kh,
                                  unsigned short* __restrict__ Vt)
{
    int idx = blockIdx.x * blockDim.x + threadIdx.x;   // N*H*L*32 threads
    int i = idx & 31;
    int l = (idx >> 5) & (LL - 1);
    int h = (idx >> 15) & (HH - 1);
    int n = idx >> 19;
    if (n >= NB) return;

    const float* row = qkv + (size_t)(n * LL + l) * (3 * DD);
    float q0 = row[h * HD + i],          q1 = row[h * HD + i + 32];
    float k0 = row[DD + h * HD + i],     k1 = row[DD + h * HD + i + 32];
    float v0 = row[2 * DD + h * HD + i], v1 = row[2 * DD + h * HD + i + 32];

    float inv = powf(10000.0f, -(float)i * (1.0f / 32.0f));
    float ang = (float)l * inv;
    float c = cosf(ang), s = sinf(ang);

    size_t hb = ((size_t)(n * HH + h) * LL + l) * HD;
    Qh[hb + i]      = f2bf_bits((q0 * c - q1 * s) * SCL);
    Qh[hb + i + 32] = f2bf_bits((q1 * c + q0 * s) * SCL);
    Kh[hb + i]      = f2bf_bits(k0 * c - k1 * s);
    Kh[hb + i + 32] = f2bf_bits(k1 * c + k0 * s);

    size_t vb = (size_t)(n * HH + h) * HD;
    Vt[(vb + i)      * LL + l] = f2bf_bits(v0);
    Vt[(vb + i + 32) * LL + l] = f2bf_bits(v1);
}

// ----------------------- attention pass A: softmax stats -------------------
// One wave per (n,h,16-row query block). Scores via 2 WMMAs per 16-key tile.

__global__ __launch_bounds__(256) void attn_stats_kernel(
    const unsigned short* __restrict__ Qh,
    const unsigned short* __restrict__ Kh,
    float* __restrict__ rowM,
    float* __restrict__ rowS)
{
    int wid  = blockIdx.x * 8 + (threadIdx.x >> 5);    // 0..8191
    int lane = threadIdx.x & 31;
    int qb = wid & 63;
    int h  = (wid >> 6) & (HH - 1);
    int n  = wid >> 10;
    int nh = n * HH + h;
    int r  = lane & 15, hi = lane >> 4;

    const unsigned int* Qu = (const unsigned int*)Qh;  // 32 uints / row
    const unsigned int* Ku = (const unsigned int*)Kh;

    size_t qrow = (size_t)nh * LL + qb * 16 + r;
    v8u aq0u, aq1u;
    #pragma unroll
    for (int v = 0; v < 8; ++v) {
        int k2u = v + 8 * hi;
        aq0u[v] = Qu[qrow * 32 + k2u];
        aq1u[v] = Qu[qrow * 32 + 16 + k2u];
    }
    v16bf aq0 = __builtin_bit_cast(v16bf, aq0u);
    v16bf aq1 = __builtin_bit_cast(v16bf, aq1u);

    float m[8], s[8];
    #pragma unroll
    for (int v = 0; v < 8; ++v) { m[v] = -1.0e30f; s[v] = 0.0f; }

    for (int jb = 0; jb < 64; ++jb) {
        size_t krow = (size_t)nh * LL + jb * 16 + r;
        v8u b0u, b1u;
        #pragma unroll
        for (int v = 0; v < 8; ++v) {
            b0u[v] = Ku[krow * 32 + v + 8 * hi];
            b1u[v] = Ku[krow * 32 + 16 + v + 8 * hi];
        }
        v8f c = {};
        c = wmma_bf16(aq0, __builtin_bit_cast(v16bf, b0u), c);
        c = wmma_bf16(aq1, __builtin_bit_cast(v16bf, b1u), c);

        #pragma unroll
        for (int v = 0; v < 8; ++v) {
            float x = c[v];
            float mt = x;
            mt = fmaxf(mt, __shfl_xor(mt, 1, 16));
            mt = fmaxf(mt, __shfl_xor(mt, 2, 16));
            mt = fmaxf(mt, __shfl_xor(mt, 4, 16));
            mt = fmaxf(mt, __shfl_xor(mt, 8, 16));
            float mn = fmaxf(m[v], mt);
            float e = __expf(x - mn);
            e += __shfl_xor(e, 1, 16);
            e += __shfl_xor(e, 2, 16);
            e += __shfl_xor(e, 4, 16);
            e += __shfl_xor(e, 8, 16);
            s[v] = s[v] * __expf(m[v] - mn) + e;
            m[v] = mn;
        }
    }
    if (r == 0) {
        size_t sb = (size_t)nh * LL + qb * 16;
        #pragma unroll
        for (int v = 0; v < 8; ++v) {
            rowM[sb + v + 8 * hi] = m[v];
            rowS[sb + v + 8 * hi] = s[v];
        }
    }
}

// ----------------- attention pass B: probs, mean, O = P@V ------------------

__global__ __launch_bounds__(256) void attn_apply_kernel(
    const unsigned short* __restrict__ Qh,
    const unsigned short* __restrict__ Kh,
    const unsigned short* __restrict__ Vt,
    const float* __restrict__ rowM,
    const float* __restrict__ rowS,
    float* __restrict__ wmean,                 // (N,L,L), pre-zeroed
    unsigned short* __restrict__ attn_bf)      // (N*L, D) bf16
{
    __shared__ unsigned int ldsP[8][256];      // per wave: 16x32 bf16 P tile

    int wave = threadIdx.x >> 5;
    int lane = threadIdx.x & 31;
    int wid  = blockIdx.x * 8 + wave;
    int qb = wid & 63;
    int h  = (wid >> 6) & (HH - 1);
    int n  = wid >> 10;
    int nh = n * HH + h;
    int r  = lane & 15, hi = lane >> 4;

    const unsigned int* Qu = (const unsigned int*)Qh;
    const unsigned int* Ku = (const unsigned int*)Kh;
    const unsigned int* Vu = (const unsigned int*)Vt;  // (N*H*hd, 512) uints

    size_t qrow = (size_t)nh * LL + qb * 16 + r;
    v8u aq0u, aq1u;
    #pragma unroll
    for (int v = 0; v < 8; ++v) {
        int k2u = v + 8 * hi;
        aq0u[v] = Qu[qrow * 32 + k2u];
        aq1u[v] = Qu[qrow * 32 + 16 + k2u];
    }
    v16bf aq0 = __builtin_bit_cast(v16bf, aq0u);
    v16bf aq1 = __builtin_bit_cast(v16bf, aq1u);

    float m[8], sinv[8];
    size_t sb = (size_t)nh * LL + qb * 16;
    #pragma unroll
    for (int v = 0; v < 8; ++v) {
        m[v]    = rowM[sb + v + 8 * hi];
        sinv[v] = 1.0f / rowS[sb + v + 8 * hi];
    }

    v8f o[4];
    #pragma unroll
    for (int nt = 0; nt < 4; ++nt) o[nt] = {};

    for (int jp = 0; jp < 32; ++jp) {          // pairs of 16-key tiles
        #pragma unroll
        for (int sub = 0; sub < 2; ++sub) {
            int jb = jp * 2 + sub;
            size_t krow = (size_t)nh * LL + jb * 16 + r;
            v8u b0u, b1u;
            #pragma unroll
            for (int v = 0; v < 8; ++v) {
                b0u[v] = Ku[krow * 32 + v + 8 * hi];
                b1u[v] = Ku[krow * 32 + 16 + v + 8 * hi];
            }
            v8f c = {};
            c = wmma_bf16(aq0, __builtin_bit_cast(v16bf, b0u), c);
            c = wmma_bf16(aq1, __builtin_bit_cast(v16bf, b1u), c);

            unsigned short* pl = (unsigned short*)ldsP[wave];
            #pragma unroll
            for (int v = 0; v < 8; ++v) {
                float p = __expf(c[v] - m[v]) * sinv[v];
                int row_l = qb * 16 + v + 8 * hi;
                int col   = jb * 16 + r;
                atomicAdd(&wmean[((size_t)n * LL + row_l) * LL + col],
                          p * (1.0f / (float)HH));
                pl[(v + 8 * hi) * 32 + sub * 16 + r] = f2bf_bits(p);
            }
        }
        // wave-private LDS tile: wait for our own DS stores before re-reading
        asm volatile("s_wait_dscnt 0" ::: "memory");

        v8u pu;
        #pragma unroll
        for (int v = 0; v < 8; ++v) {
            int k2u = v + 8 * hi;
            pu[v] = ldsP[wave][r * 16 + k2u];
        }
        v16bf ap = __builtin_bit_cast(v16bf, pu);

        #pragma unroll
        for (int nt = 0; nt < 4; ++nt) {
            v8u bu;
            #pragma unroll
            for (int v = 0; v < 8; ++v)
                bu[v] = Vu[((size_t)nh * HD + nt * 16 + r) * 512
                           + jp * 16 + v + 8 * hi];
            o[nt] = wmma_bf16(ap, __builtin_bit_cast(v16bf, bu), o[nt]);
        }
    }

    #pragma unroll
    for (int nt = 0; nt < 4; ++nt)
        #pragma unroll
        for (int v = 0; v < 8; ++v) {
            size_t row = (size_t)n * LL + qb * 16 + v + 8 * hi;
            attn_bf[row * DD + h * HD + nt * 16 + r] = f2bf_bits(o[nt][v]);
        }
}

// ------------------------------ launcher -----------------------------------

extern "C" void kernel_launch(void* const* d_in, const int* in_sizes, int n_in,
                              void* d_out, int out_size, void* d_ws, size_t ws_size,
                              hipStream_t stream) {
    const float* x     = (const float*)d_in[0];   // (8,1024,1024)
    const float* W_in  = (const float*)d_in[1];   // (3072,1024)
    const float* b_in  = (const float*)d_in[2];   // (3072)
    const float* W_out = (const float*)d_in[3];   // (1024,1024)
    const float* b_out = (const float*)d_in[4];   // (1024)

    float* out   = (float*)d_out;                                    // 8M f32
    float* wmean = (float*)d_out + (size_t)NB * LL * LL;             // 8M f32

    char* ws = (char*)d_ws;
    unsigned short* x_bf    = (unsigned short*)(ws);                 // 16 MB
    unsigned short* Win_bf  = (unsigned short*)(ws + 16777216u);     //  6 MB
    unsigned short* Wout_bf = (unsigned short*)(ws + 23068672u);     //  2 MB
    float*          qkv     = (float*)        (ws + 25165824u);      // 96 MB
    unsigned short* Qh      = (unsigned short*)(ws + 125829120u);    // 16 MB
    unsigned short* Kh      = (unsigned short*)(ws + 142606336u);    // 16 MB
    unsigned short* Vt      = (unsigned short*)(ws + 159383552u);    // 16 MB
    float*          rowM    = (float*)        (ws + 176160768u);     // 0.5 MB
    float*          rowS    = (float*)        (ws + 176685056u);     // 0.5 MB
    unsigned short* attn_bf = (unsigned short*)(ws + 177209344u);    // 16 MB

    const size_t nX = (size_t)NB * LL * DD;          // 8,388,608
    const size_t nWin = (size_t)3 * DD * DD;         // 3,145,728
    const size_t nWout = (size_t)DD * DD;            // 1,048,576

    cvt_f32_bf16_kernel<<<(nX + 255) / 256, 256, 0, stream>>>(x, x_bf, nX);
    cvt_f32_bf16_kernel<<<(nWin + 255) / 256, 256, 0, stream>>>(W_in, Win_bf, nWin);
    cvt_f32_bf16_kernel<<<(nWout + 255) / 256, 256, 0, stream>>>(W_out, Wout_bf, nWout);

    // qkv = x @ W_in^T + b_in   (M=8192, N=3072, K=1024)
    gemm_bf16_bias_kernel<<<dim3(3 * DD / 128, NB * LL / 128), 256, 0, stream>>>(
        x_bf, Win_bf, b_in, qkv, NB * LL, 3 * DD, DD);

    rope_split_kernel<<<(NB * HH * LL * 32) / 256, 256, 0, stream>>>(qkv, Qh, Kh, Vt);

    zero_f32_kernel<<<((size_t)NB * LL * LL + 255) / 256, 256, 0, stream>>>(
        wmean, (size_t)NB * LL * LL);

    attn_stats_kernel<<<NB * HH * (LL / 16) / 8, 256, 0, stream>>>(Qh, Kh, rowM, rowS);

    attn_apply_kernel<<<NB * HH * (LL / 16) / 8, 256, 0, stream>>>(
        Qh, Kh, Vt, rowM, rowS, wmean, attn_bf);

    // out = attn @ W_out^T + b_out   (M=8192, N=1024, K=1024)
    gemm_bf16_bias_kernel<<<dim3(DD / 128, NB * LL / 128), 256, 0, stream>>>(
        attn_bf, Wout_bf, b_out, out, NB * LL, DD, DD);
}
